// PNALayer_19567871000704
// MI455X (gfx1250) — compile-verified, matrix operand order checked by reading, not run on hardware
//
#include <hip/hip_runtime.h>
#include <hip/hip_fp16.h>
#include <stdint.h>
#include <math.h>

#define HID 128
#define AVG_DEG_LOG 2.8332133440562162f  // log(17)

typedef __attribute__((ext_vector_type(16))) _Float16 v16h;
typedef __attribute__((ext_vector_type(2)))  __fp16   v2hp;  // matches cvt_pkrtz return
typedef __attribute__((ext_vector_type(8)))  float    v8f;

// ---------------------------------------------------------------------------
// WMMA helpers (v_wmma_f32_16x16x32_f16, wave32)
// 16-bit A/B fragment per ISA 7.12.2: lane = half*16 + i (i = M for A, N for B^T)
// element pair (2v,2v+1) holds K = k0,k0+1, k0 = (v<4?0:16)+half*8+(v%4)*2
// C tile: lane col = lane&15, rows = vgpr_r + 8*(lane>>4)
// ---------------------------------------------------------------------------
__device__ inline v16h frag_ld(const _Float16* base, int rstride) {
  int lane = threadIdx.x & 31;
  int i = lane & 15, half = lane >> 4;
  v16h f;
#pragma unroll
  for (int v = 0; v < 8; ++v) {
    int k0 = ((v & 4) << 2) + (half << 3) + ((v & 3) << 1);
    const _Float16* p = base + (size_t)i * rstride + k0;
    f[2 * v]     = p[0];
    f[2 * v + 1] = p[1];
  }
  return f;
}

__device__ inline v8f wmma_acc(v16h a, v16h b, v8f c) {
  return __builtin_amdgcn_wmma_f32_16x16x32_f16(
      false, a, false, b, (short)0, c, false, false);
}

// pack two f32 to packed f16 (v_cvt_pk_rtz_f16_f32) and store to LDS tile slot
__device__ inline void pack_store(_Float16* slot, float a, float b) {
  *(v2hp*)slot = __builtin_amdgcn_cvt_pkrtz(a, b);
}

// gfx1250 async gather: memory -> LDS, per-lane 4 bytes, tracked by ASYNCcnt
__device__ inline void async_gather_b32(uint32_t lds_byte_off, const void* gaddr) {
  asm volatile("global_load_async_to_lds_b32 %0, %1, off"
               :: "v"(lds_byte_off), "v"((uint64_t)(uintptr_t)gaddr)
               : "memory");
}
__device__ inline void wait_async0() {
  asm volatile("s_wait_asynccnt 0" ::: "memory");
}
__device__ inline uint32_t lds_off_u32(const void* p) {
  return (uint32_t)(uintptr_t)p;  // low 32 bits of generic LDS addr = LDS offset
}

__device__ inline void atomicMaxF(float* addr, float v) {
  if (v >= 0.f) atomicMax((int*)addr, __float_as_int(v));
  else          atomicMin((unsigned int*)addr, __float_as_uint(v));
}
__device__ inline void atomicMinF(float* addr, float v) {
  if (v >= 0.f) atomicMin((int*)addr, __float_as_int(v));
  else          atomicMax((unsigned int*)addr, __float_as_uint(v));
}

// ---------------------------------------------------------------------------
__global__ void kinit(float* __restrict__ sumb, float* __restrict__ sqb,
                      float* __restrict__ maxb, float* __restrict__ minb,
                      float* __restrict__ deg, float* __restrict__ bnacc,
                      size_t NH, int N) {
  size_t i = (size_t)blockIdx.x * blockDim.x + threadIdx.x;
  if (i < NH) {
    sumb[i] = 0.f; sqb[i] = 0.f;
    maxb[i] = -3.402823466e+38f;
    minb[i] =  3.402823466e+38f;
  }
  if (i < (size_t)N) deg[i] = 0.f;
  if (i < 256)       bnacc[i] = 0.f;
}

// transpose-convert weight [K,128] f32 -> [128,K] f16
__global__ void kw_convT(const float* __restrict__ W, _Float16* __restrict__ Wt, int K) {
  int i = blockIdx.x * blockDim.x + threadIdx.x;
  if (i >= K * 128) return;
  int k = i >> 7, n = i & 127;
  Wt[(size_t)n * K + k] = (_Float16)W[i];
}

// elementwise fp32 -> f16
__global__ void kcvt16(const float* __restrict__ a, _Float16* __restrict__ o, size_t n) {
  size_t i = (size_t)blockIdx.x * blockDim.x + threadIdx.x;
  if (i < n) o[i] = (_Float16)a[i];
}

__global__ void kdeg(const int* __restrict__ dst, float* __restrict__ deg, int E) {
  int e = blockIdx.x * blockDim.x + threadIdx.x;
  if (e < E) atomicAdd(&deg[dst[e]], 1.f);
}

// ---------------------------------------------------------------------------
// K1: e_enc = edge_attr @ We + be  -> f16   (E x 128, K=128)
// ---------------------------------------------------------------------------
__global__ __launch_bounds__(256) void k1_eenc(
    const float* __restrict__ edge_attr, const _Float16* __restrict__ We16t,
    const float* __restrict__ be, _Float16* __restrict__ eenc16) {
  __shared__ _Float16 atile[16 * 32];
  int e0 = blockIdx.x * 16;
  int tid = threadIdx.x;
  int nbase = (tid >> 5) * 16;
  int r = tid >> 4, kk2 = tid & 15;           // tid == r*16 + kk2
  v8f c = {};
  for (int kb = 0; kb < 128; kb += 32) {
    __syncthreads();
    {
      const float* p = edge_attr + (size_t)(e0 + r) * HID + kb + kk2 * 2;
      float2 t = *(const float2*)p;
      pack_store(&atile[tid * 2], t.x, t.y);
    }
    __syncthreads();
    v16h a = frag_ld(atile, 32);
    v16h b = frag_ld(We16t + (size_t)nbase * 128 + kb, 128);
    c = wmma_acc(a, b, c);
  }
  int lane = tid & 31;
  int col = nbase + (lane & 15);
  int rb = (lane >> 4) * 8;
  float bias = be[col];
#pragma unroll
  for (int rr = 0; rr < 8; ++rr)
    eenc16[(size_t)(e0 + rb + rr) * HID + col] = (_Float16)(c[rr] + bias);
}

// ---------------------------------------------------------------------------
// K2: h = [x16[dst] | x16[src] | e_enc16] @ Wpre + bpre (K=384); atomic scatter
//     A-tiles gathered memory->LDS with global_load_async_to_lds_b32.
// ---------------------------------------------------------------------------
__global__ __launch_bounds__(256) void k2_msg(
    const _Float16* __restrict__ x16, const _Float16* __restrict__ eenc16,
    const _Float16* __restrict__ Wpre16t, const float* __restrict__ bpre,
    const int* __restrict__ src, const int* __restrict__ dst,
    float* __restrict__ sumb, float* __restrict__ sqb,
    float* __restrict__ maxb, float* __restrict__ minb) {
  __shared__ _Float16 atile[16 * 32];
  __shared__ int sdst[16], ssrc[16];
  int e0 = blockIdx.x * 16;
  int tid = threadIdx.x;
  if (tid < 16)      sdst[tid]      = dst[e0 + tid];
  else if (tid < 32) ssrc[tid - 16] = src[e0 + tid - 16];
  int nbase = (tid >> 5) * 16;
  int r = tid >> 4, kk2 = tid & 15;
  uint32_t ldsb = lds_off_u32(atile) + tid * 4;
  v8f c = {};
  for (int kb = 0; kb < 384; kb += 32) {
    __syncthreads();
    {
      int fo = (kb & 127) + kk2 * 2;
      const _Float16* rowp;
      if (kb < 128)      rowp = x16    + (size_t)sdst[r] * HID + fo;
      else if (kb < 256) rowp = x16    + (size_t)ssrc[r] * HID + fo;
      else               rowp = eenc16 + (size_t)(e0 + r) * HID + fo;
      async_gather_b32(ldsb, rowp);
      wait_async0();
    }
    __syncthreads();
    v16h a = frag_ld(atile, 32);
    v16h b = frag_ld(Wpre16t + (size_t)nbase * 384 + kb, 384);
    c = wmma_acc(a, b, c);
  }
  int lane = tid & 31;
  int col = nbase + (lane & 15);
  int rb = (lane >> 4) * 8;
  float bias = bpre[col];
#pragma unroll
  for (int rr = 0; rr < 8; ++rr) {
    float h = c[rr] + bias;
    size_t idx = (size_t)sdst[rb + rr] * HID + col;
    atomicAdd(&sumb[idx], h);
    atomicAdd(&sqb[idx], h * h);
    atomicMaxF(&maxb[idx], h);
    atomicMinF(&minb[idx], h);
  }
}

// ---------------------------------------------------------------------------
// K3: out2 = ([x | 12H scaled aggregates] @ Wpost + bpost) @ Wlin + blin
// ---------------------------------------------------------------------------
__global__ __launch_bounds__(256) void k3_post(
    const float* __restrict__ x,
    const float* __restrict__ sumb, const float* __restrict__ sqb,
    const float* __restrict__ maxb, const float* __restrict__ minb,
    const float* __restrict__ deg,
    const _Float16* __restrict__ Wpost16t, const float* __restrict__ bpost,
    const _Float16* __restrict__ Wlin16t, const float* __restrict__ blin,
    float* __restrict__ out2, float* __restrict__ bnacc) {
  __shared__ _Float16 atile[16 * 32];
  __shared__ _Float16 hbuf[16 * 128];
  int n0 = blockIdx.x * 16;
  int tid = threadIdx.x;
  int nbase = (tid >> 5) * 16;
  int r = tid >> 4, kk2 = tid & 15;
  int node = n0 + r;
  v8f c = {};
  for (int kb = 0; kb < 1664; kb += 32) {
    __syncthreads();
    {
      int f = kb + kk2 * 2;          // pair (f, f+1), never straddles 128-block
      float v0, v1;
      if (f < 128) {
        float2 t = *(const float2*)(x + (size_t)node * HID + f);
        v0 = t.x; v1 = t.y;
      } else {
        int j  = f - 128;
        int sg = j / 512;            // 0 raw, 1 *logd/avg, 2 *avg/logd
        int jj = j & 511;
        int g  = jj >> 7;            // 0 mean, 1 max, 2 min, 3 std
        int cc = jj & 127;
        float d  = deg[node];
        float dc = fmaxf(d, 1.f);
        float rdc = 1.f / dc;
        size_t idx = (size_t)node * HID + cc;
        if (g == 0) {
          v0 = sumb[idx] * rdc; v1 = sumb[idx + 1] * rdc;
        } else if (g == 1) {
          bool he = d > 0.f;
          v0 = he ? maxb[idx] : 0.f; v1 = he ? maxb[idx + 1] : 0.f;
        } else if (g == 2) {
          bool he = d > 0.f;
          v0 = he ? minb[idx] : 0.f; v1 = he ? minb[idx + 1] : 0.f;
        } else {
          float m0 = sumb[idx] * rdc,     m1 = sumb[idx + 1] * rdc;
          float q0 = sqb[idx] * rdc,      q1 = sqb[idx + 1] * rdc;
          v0 = sqrtf(fmaxf(q0 - m0 * m0, 0.f) + 1e-5f);
          v1 = sqrtf(fmaxf(q1 - m1 * m1, 0.f) + 1e-5f);
        }
        if (sg == 1) {
          float s = logf(dc + 1.f) * (1.f / AVG_DEG_LOG); v0 *= s; v1 *= s;
        } else if (sg == 2) {
          float s = AVG_DEG_LOG / logf(dc + 1.f); v0 *= s; v1 *= s;
        }
      }
      pack_store(&atile[tid * 2], v0, v1);
    }
    __syncthreads();
    v16h a = frag_ld(atile, 32);
    v16h b = frag_ld(Wpost16t + (size_t)nbase * 1664 + kb, 1664);
    c = wmma_acc(a, b, c);
  }
  int lane = tid & 31;
  int col = nbase + (lane & 15);
  int rb = (lane >> 4) * 8;
  float bias = bpost[col];
  __syncthreads();
#pragma unroll
  for (int rr = 0; rr < 8; ++rr)
    hbuf[(rb + rr) * 128 + col] = (_Float16)(c[rr] + bias);
  __syncthreads();
  v8f c2 = {};
#pragma unroll
  for (int kb = 0; kb < 128; kb += 32) {
    v16h a = frag_ld(hbuf + kb, 128);
    v16h b = frag_ld(Wlin16t + (size_t)nbase * 128 + kb, 128);
    c2 = wmma_acc(a, b, c2);
  }
  float bias2 = blin[col];
  float ps = 0.f, pq = 0.f;
#pragma unroll
  for (int rr = 0; rr < 8; ++rr) {
    float v = c2[rr] + bias2;
    out2[(size_t)(n0 + rb + rr) * HID + col] = v;
    ps += v; pq += v * v;
  }
  atomicAdd(&bnacc[col], ps);
  atomicAdd(&bnacc[128 + col], pq);
}

__global__ void k4_bnstats(const float* __restrict__ bnacc,
                           float* __restrict__ bnstat, float invN) {
  int c = threadIdx.x;
  float mu  = bnacc[c] * invN;
  float var = bnacc[128 + c] * invN - mu * mu;
  bnstat[c]       = mu;
  bnstat[128 + c] = rsqrtf(var + 1e-5f);
}

__global__ void k5_xnew(const float* __restrict__ x, const float* __restrict__ out2,
                        const float* __restrict__ bnstat,
                        const float* __restrict__ bn_g, const float* __restrict__ bn_b,
                        float* __restrict__ xout, _Float16* __restrict__ xnew16,
                        size_t total) {
  size_t i = (size_t)blockIdx.x * blockDim.x + threadIdx.x;
  if (i >= total) return;
  int c = (int)(i & 127);
  float bn = (out2[i] - bnstat[c]) * bnstat[128 + c] * bn_g[c] + bn_b[c];
  float v = (x[i] + fmaxf(bn, 0.f)) * 0.5f;
  xout[i]   = v;
  xnew16[i] = (_Float16)v;
}

// ---------------------------------------------------------------------------
// K6: e_new = edge_attr + 0.5*(relu([xn[src]|xn[dst]|edge_attr]@W1+b1)@W2+b2)
//     x_new gathers use async-to-LDS; edge_attr segment converts f32 pairs.
// ---------------------------------------------------------------------------
__global__ __launch_bounds__(256) void k6_edge(
    const _Float16* __restrict__ xnew16, const float* __restrict__ edge_attr,
    const _Float16* __restrict__ W116t, const float* __restrict__ b1,
    const _Float16* __restrict__ W216t, const float* __restrict__ b2,
    const int* __restrict__ src, const int* __restrict__ dst,
    float* __restrict__ eout) {
  __shared__ _Float16 atile[16 * 32];
  __shared__ _Float16 tbuf[16 * 128];
  __shared__ int sdst[16], ssrc[16];
  int e0 = blockIdx.x * 16;
  int tid = threadIdx.x;
  if (tid < 16)      ssrc[tid]      = src[e0 + tid];
  else if (tid < 32) sdst[tid - 16] = dst[e0 + tid - 16];
  int nbase = (tid >> 5) * 16;
  int r = tid >> 4, kk2 = tid & 15;
  uint32_t ldsb = lds_off_u32(atile) + tid * 4;
  v8f c = {};
  for (int kb = 0; kb < 384; kb += 32) {
    __syncthreads();
    if (kb < 256) {
      int fo = (kb & 127) + kk2 * 2;
      const _Float16* rowp = (kb < 128)
          ? xnew16 + (size_t)ssrc[r] * HID + fo
          : xnew16 + (size_t)sdst[r] * HID + fo;
      async_gather_b32(ldsb, rowp);
      wait_async0();
    } else {
      const float* p = edge_attr + (size_t)(e0 + r) * HID + (kb - 256) + kk2 * 2;
      float2 t = *(const float2*)p;
      pack_store(&atile[tid * 2], t.x, t.y);
    }
    __syncthreads();
    v16h a = frag_ld(atile, 32);
    v16h b = frag_ld(W116t + (size_t)nbase * 384 + kb, 384);
    c = wmma_acc(a, b, c);
  }
  int lane = tid & 31;
  int col = nbase + (lane & 15);
  int rb = (lane >> 4) * 8;
  float bias = b1[col];
  __syncthreads();
#pragma unroll
  for (int rr = 0; rr < 8; ++rr)
    tbuf[(rb + rr) * 128 + col] = (_Float16)fmaxf(c[rr] + bias, 0.f);
  __syncthreads();
  v8f c2 = {};
#pragma unroll
  for (int kb = 0; kb < 128; kb += 32) {
    v16h a = frag_ld(tbuf + kb, 128);
    v16h b = frag_ld(W216t + (size_t)nbase * 128 + kb, 128);
    c2 = wmma_acc(a, b, c2);
  }
  float bias2 = b2[col];
#pragma unroll
  for (int rr = 0; rr < 8; ++rr) {
    size_t idx = (size_t)(e0 + rb + rr) * HID + col;
    eout[idx] = edge_attr[idx] + 0.5f * (c2[rr] + bias2);
  }
}

// ---------------------------------------------------------------------------
extern "C" void kernel_launch(void* const* d_in, const int* in_sizes, int n_in,
                              void* d_out, int out_size, void* d_ws, size_t ws_size,
                              hipStream_t stream) {
  (void)n_in; (void)out_size; (void)ws_size;
  const float* x         = (const float*)d_in[0];
  const float* edge_attr = (const float*)d_in[1];
  const float* We    = (const float*)d_in[2];
  const float* be    = (const float*)d_in[3];
  const float* Wpre  = (const float*)d_in[4];
  const float* bpre  = (const float*)d_in[5];
  const float* Wpost = (const float*)d_in[6];
  const float* bpost = (const float*)d_in[7];
  const float* Wlin  = (const float*)d_in[8];
  const float* blin  = (const float*)d_in[9];
  const float* bn_g  = (const float*)d_in[10];
  const float* bn_b  = (const float*)d_in[11];
  const float* W1    = (const float*)d_in[12];
  const float* b1    = (const float*)d_in[13];
  const float* W2    = (const float*)d_in[14];
  const float* b2    = (const float*)d_in[15];
  const int*   ei    = (const int*)d_in[16];

  const int N = in_sizes[0] / HID;   // 50000 (multiple of 16)
  const int E = in_sizes[1] / HID;   // 600000 (multiple of 16)
  const int* srcp = ei;
  const int* dstp = ei + E;

  // ---- workspace carve-up (256B aligned) ----
  uint8_t* base = (uint8_t*)d_ws;
  size_t off = 0;
  auto carve = [&](size_t bytes) -> void* {
    void* p = base + off;
    off += (bytes + 255) & ~(size_t)255;
    return p;
  };
  _Float16* We16t    = (_Float16*)carve((size_t)128 * 128 * 2);
  _Float16* Wpre16t  = (_Float16*)carve((size_t)128 * 384 * 2);
  _Float16* Wpost16t = (_Float16*)carve((size_t)128 * 1664 * 2);
  _Float16* Wlin16t  = (_Float16*)carve((size_t)128 * 128 * 2);
  _Float16* W116t    = (_Float16*)carve((size_t)128 * 384 * 2);
  _Float16* W216t    = (_Float16*)carve((size_t)128 * 128 * 2);
  float* sumb  = (float*)carve((size_t)N * HID * 4);
  float* sqb   = (float*)carve((size_t)N * HID * 4);
  float* maxb  = (float*)carve((size_t)N * HID * 4);
  float* minb  = (float*)carve((size_t)N * HID * 4);
  float* deg   = (float*)carve((size_t)N * 4);
  float* bnacc = (float*)carve(256 * 4);
  float* bnstat= (float*)carve(256 * 4);
  _Float16* x16 = (_Float16*)carve((size_t)N * HID * 2);
  uint8_t* big = (uint8_t*)carve((size_t)E * HID * 2);   // 153.6 MB region
  _Float16* eenc16 = (_Float16*)big;                     // live K1 -> K2
  float*    out2   = (float*)big;                        // alias, live K3 -> K5
  _Float16* xnew16 = (_Float16*)(big + (size_t)N * HID * 4); // alias, K5 -> K6

  size_t NH = (size_t)N * HID;

  kinit<<<(int)((NH + 255) / 256), 256, 0, stream>>>(sumb, sqb, maxb, minb, deg, bnacc, NH, N);
  kw_convT<<<(128 * 128 + 255) / 256, 256, 0, stream>>>(We, We16t, 128);
  kw_convT<<<(384 * 128 + 255) / 256, 256, 0, stream>>>(Wpre, Wpre16t, 384);
  kw_convT<<<(1664 * 128 + 255) / 256, 256, 0, stream>>>(Wpost, Wpost16t, 1664);
  kw_convT<<<(128 * 128 + 255) / 256, 256, 0, stream>>>(Wlin, Wlin16t, 128);
  kw_convT<<<(384 * 128 + 255) / 256, 256, 0, stream>>>(W1, W116t, 384);
  kw_convT<<<(128 * 128 + 255) / 256, 256, 0, stream>>>(W2, W216t, 128);
  kcvt16<<<(int)((NH + 255) / 256), 256, 0, stream>>>(x, x16, NH);
  kdeg<<<(E + 255) / 256, 256, 0, stream>>>(dstp, deg, E);

  k1_eenc<<<E / 16, 256, 0, stream>>>(edge_attr, We16t, be, eenc16);
  k2_msg <<<E / 16, 256, 0, stream>>>(x16, eenc16, Wpre16t, bpre, srcp, dstp,
                                      sumb, sqb, maxb, minb);
  k3_post<<<N / 16, 256, 0, stream>>>(x, sumb, sqb, maxb, minb, deg,
                                      Wpost16t, bpost, Wlin16t, blin, out2, bnacc);
  k4_bnstats<<<1, 128, 0, stream>>>(bnacc, bnstat, 1.f / (float)N);

  float* xout = (float*)d_out;
  float* eout = xout + NH;
  k5_xnew<<<(int)((NH + 255) / 256), 256, 0, stream>>>(x, out2, bnstat, bn_g, bn_b,
                                                       xout, xnew16, NH);
  k6_edge<<<E / 16, 256, 0, stream>>>(xnew16, edge_attr, W116t, b1, W216t, b2,
                                      srcp, dstp, eout);
}
